// GraphEdgeActionGNN_63900523429920
// MI455X (gfx1250) — compile-verified, matrix operand order checked by reading, not run on hardware
//
#include <hip/hip_runtime.h>

#define NT      262144          // total nodes (B * N_NODES)
#define NNODES  256
#define EPG     2048            // edges per graph
#define NGRAPH  1024
#define OUTW    32641           // 32640 triu edges + 1 exit action
#define NTF     (NT * 128)
#define SWP     136             // padded LDS row stride (16B aligned, bank-skewed)

typedef __bf16 bf16_t;
typedef __attribute__((ext_vector_type(16))) __bf16 v16bf;
typedef __attribute__((ext_vector_type(8)))  __bf16 v8bf;
typedef __attribute__((ext_vector_type(2)))  __bf16 v2bf;
typedef __attribute__((ext_vector_type(8)))  float  v8f;
typedef __attribute__((ext_vector_type(4)))  float  v4f;
typedef __attribute__((ext_vector_type(2)))  float  v2f;
typedef __attribute__((ext_vector_type(4)))  unsigned v4u;
typedef __attribute__((ext_vector_type(8)))  unsigned v8u;

union FragBF { v16bf v; v8bf h[2]; };

// ---------------------------------------------------------------- utilities
__global__ void k_f2b(const float* __restrict__ in, bf16_t* __restrict__ out, int n) {
    int i = blockIdx.x * 256 + threadIdx.x;
    if (i < n) out[i] = (bf16_t)in[i];
}

__global__ void k_embed(const int* __restrict__ nf, const float* __restrict__ emb,
                        float* __restrict__ X) {
    int i = blockIdx.x * 256 + threadIdx.x;      // over NT*128
    int n = i >> 7, f = i & 127;
    X[i] = emb[nf[n] * 128 + f];
}

// ------------------------------------------------- per-graph LDS segment-sum
// h0 = x + segment_sum(x[src], dst), emitted as bf16 (feeds GEMMs only)
__global__ __launch_bounds__(1024) void k_aggregate(
    const float* __restrict__ X, const int* __restrict__ src,
    const int* __restrict__ dst, bf16_t* __restrict__ Hb) {
    __shared__ float sAgg[NNODES * 128];         // 128 KB of the 320 KB WGP LDS
    int g = blockIdx.x, tid = threadIdx.x;
    for (int i = tid; i < NNODES * 32; i += 1024)
        ((v4f*)sAgg)[i] = (v4f){0.f, 0.f, 0.f, 0.f};
    __syncthreads();
    int f = tid & 127;
    int es = tid >> 7;                           // 8 edges in flight
    int ebase = g * EPG, nbase = g * NNODES;
    for (int e0 = es; e0 < EPG; e0 += 8) {
        int e  = ebase + e0;
        int sN = src[e], dN = dst[e];
        float val = X[(size_t)sN * 128 + f];
        atomicAdd(&sAgg[(dN - nbase) * 128 + f], val);   // ds_add_f32
    }
    __syncthreads();
    size_t base = (size_t)nbase * 128;
    const v2f* Xp = (const v2f*)(X + base);
    v2bf* Hp = (v2bf*)(Hb + base);
    for (int i = tid; i < NNODES * 64; i += 1024) {
        v2f xv = Xp[i], av = ((const v2f*)sAgg)[i];
        v2bf pk; pk.x = (bf16_t)(xv.x + av.x); pk.y = (bf16_t)(xv.y + av.y);
        Hp[i] = pk;                               // packed b32 store
    }
}

// -------------------------------------------------------------- WMMA GEMM
// out = act(A[rows x 128] @ W[128 x 128] + bias); A bf16, W bf16 row-major.
// 256 threads = 8 waves; each wave owns a 32x128 strip (2 A-strips reuse B).
template<int RELU, int HASF, int HASB>
__global__ __launch_bounds__(256) void k_gemm128(
    const bf16_t* __restrict__ A, const bf16_t* __restrict__ W,
    const float* __restrict__ bias, float* __restrict__ outF,
    bf16_t* __restrict__ outB, int rows) {
    __shared__ bf16_t sWt[128 * SWP];            // W transposed + bank-skew pad
    int tid = threadIdx.x;
    for (int i = tid; i < 128 * 128; i += 256) {
        int k = i >> 7, n = i & 127;
        sWt[n * SWP + k] = W[i];                 // coalesced read, skewed write
    }
    __syncthreads();
    int wave = tid >> 5, lane = tid & 31;
    int mbase = blockIdx.x * 256 + wave * 32;
    int lrow = lane & 15;
    int koff = (lane >> 4) << 3;                 // K half-split across half-waves
    v16bf a[2][4];
    for (int s = 0; s < 2; ++s)
        for (int k4 = 0; k4 < 4; ++k4) {
            const bf16_t* p = A + (size_t)(mbase + s * 16 + lrow) * 128 + k4 * 32 + koff;
            FragBF u; u.h[0] = *(const v8bf*)p; u.h[1] = *(const v8bf*)(p + 16);
            a[s][k4] = u.v;
        }
    for (int t = 0; t < 8; ++t) {                // 8 N-tiles of 16
        v16bf b[4];
        for (int k4 = 0; k4 < 4; ++k4) {
            const bf16_t* q = sWt + (t * 16 + lrow) * SWP + k4 * 32 + koff;
            FragBF u; u.h[0] = *(const v8bf*)q; u.h[1] = *(const v8bf*)(q + 16);
            b[k4] = u.v;
        }
        int nc = t * 16 + lrow;
        float bv = bias[nc];
        for (int s = 0; s < 2; ++s) {
            v8f c = {};
            for (int k4 = 0; k4 < 4; ++k4)
                c = __builtin_amdgcn_wmma_f32_16x16x32_bf16(false, a[s][k4], false, b[k4],
                                                            (short)0, c, false, false);
            for (int r = 0; r < 8; ++r) {
                int m = mbase + s * 16 + r + koff;    // koff == 8*(lane>>4)
                float v = c[r] + bv;
                if (RELU) v = fmaxf(v, 0.f);
                if (HASF) outF[(size_t)m * 128 + nc] = v;
                if (HASB) outB[(size_t)m * 128 + nc] = (bf16_t)v;
            }
        }
    }
    (void)rows;
}

// ------------------------------------------------- LayerNorm (+res, +relu)
template<int RES, int RELU, int HASF, int HASB>
__global__ __launch_bounds__(256) void k_ln(
    const float* __restrict__ in, const float* __restrict__ res,
    const float* __restrict__ g, const float* __restrict__ b,
    float* __restrict__ outF, bf16_t* __restrict__ outB, int rows) {
    int wave = threadIdx.x >> 5, lane = threadIdx.x & 31;
    int row = blockIdx.x * 8 + wave;
    if (row >= rows) return;
    size_t base = (size_t)row * 128;
    float v[4], s = 0.f;
    for (int i = 0; i < 4; ++i) {
        int c = lane + i * 32;
        float x = in[base + c];
        if (RES) x += res[base + c];
        v[i] = x; s += x;
    }
    for (int o = 16; o > 0; o >>= 1) s += __shfl_xor(s, o, 32);
    float mu = s * (1.f / 128.f), q = 0.f;
    for (int i = 0; i < 4; ++i) { float d = v[i] - mu; q += d * d; }
    for (int o = 16; o > 0; o >>= 1) q += __shfl_xor(q, o, 32);
    float rstd = rsqrtf(q * (1.f / 128.f) + 1e-5f);
    for (int i = 0; i < 4; ++i) {
        int c = lane + i * 32;
        float y = (v[i] - mu) * rstd * g[c] + b[c];
        if (RELU) y = fmaxf(y, 0.f);
        if (HASF) outF[base + c] = y;
        if (HASB) outB[base + c] = (bf16_t)y;
    }
}

// ------------------------------------------------------------- mean pooling
__global__ void k_meanpool(const float* __restrict__ X, float* __restrict__ means,
                           bf16_t* __restrict__ meansb) {
    int g = blockIdx.x, f = threadIdx.x;         // 128 threads
    const float* p = X + (size_t)g * NNODES * 128 + f;
    float s = 0.f;
    for (int n = 0; n < NNODES; ++n) s += p[n * 128];
    float m = s * (1.f / NNODES);
    means[(size_t)g * 128 + f]  = m;
    meansb[(size_t)g * 128 + f] = (bf16_t)m;
}

// ------------------------------------------------------------ exit head dot
__global__ __launch_bounds__(256) void k_exit_dot(
    const float* __restrict__ ex, const float* __restrict__ w,
    const float* __restrict__ b, float* __restrict__ out) {
    int wave = threadIdx.x >> 5, lane = threadIdx.x & 31;
    int g = blockIdx.x * 8 + wave;
    float s = 0.f;
    for (int i = 0; i < 4; ++i) {
        int c = lane + i * 32;
        s += fmaxf(ex[(size_t)g * 128 + c], 0.f) * w[c];
    }
    for (int o = 16; o > 0; o >>= 1) s += __shfl_xor(s, o, 32);
    if (lane == 0) out[(size_t)g * OUTW + 32640] = s + b[0];
}

// -------------------------------------- per-graph pairwise dots, triu only
// TDM stages the 256x128 bf16 graph block into LDS; the DMA's pad feature
// (4 dwords every 64 dwords) produces the 136-element bank-skewed row stride.
__global__ __launch_bounds__(256) void k_dots(const bf16_t* __restrict__ Xb,
                                              float* __restrict__ out) {
    __shared__ bf16_t sX[NNODES * SWP];          // 68 KB padded
    __shared__ int rowBase[NNODES];              // triu row-base LUT
    int g = blockIdx.x, tid = threadIdx.x;
    int wave = tid >> 5, lane = tid & 31;
    if (tid < NNODES)                            // idx(m,n) = rowBase[m] + n
        rowBase[tid] = tid * 255 - (tid * (tid - 1)) / 2 - tid - 1;
    if (wave == 0) {
        const bf16_t* src = Xb + (size_t)g * NNODES * 128;
        unsigned long long ga = (unsigned long long)(const void*)src;
        unsigned lds_off = (unsigned)(unsigned long long)(uintptr_t)&sX[0];
        // D# group0: count=1, lds_addr, global_addr[56:0], type=2
        v4u g0 = { 1u, lds_off, (unsigned)ga,
                   (unsigned)(ga >> 32) | (2u << 30) };
        // D# group1: data_size=2B, pad_enable, pad_interval=64dw, pad_amount=4dw,
        // tensor_dim0 = tile_dim0 = 32768 elems (1D), stride 32768, tensor_dim1=1
        v8u g1 = { (1u << 16) | (1u << 20) | (5u << 22) | (3u << 25),
                   0x8000u << 16,          // tensor_dim0[15:0] in bits 63:48
                   1u << 16,               // tensor_dim0[31:16]=0 | tensor_dim1[15:0]=1
                   0x8000u << 16,          // tensor_dim1 hi =0 | tile_dim0=32768
                   0u,                     // tile_dim1=0, tile_dim2=0
                   32768u,                 // tensor_dim0_stride lo
                   0u, 0u };
        v4u gz = { 0u, 0u, 0u, 0u };
        asm volatile("tensor_load_to_lds %0, %1, %2, %3"
                     :: "s"(g0), "s"(g1), "s"(gz), "s"(gz) : "memory");
#if defined(__HIP_DEVICE_COMPILE__) && __has_builtin(__builtin_amdgcn_s_wait_tensorcnt)
        __builtin_amdgcn_s_wait_tensorcnt(0);
#else
        asm volatile("s_wait_tensorcnt 0x0" ::: "memory");
#endif
    }
    __syncthreads();
    int lrow = lane & 15, koff = (lane >> 4) << 3;
    float* og = out + (size_t)g * OUTW;

    // 120 strictly-upper 16x16 tiles: unpredicated straight-line stores
    for (int l = wave; l < 120; l += 8) {
        int ti = 0, rem = l;
        while (rem >= 15 - ti) { rem -= 15 - ti; ++ti; }
        int tj = ti + 1 + rem;
        v8f c = {};
        for (int k4 = 0; k4 < 4; ++k4) {
            FragBF ua, ub;
            const bf16_t* pa = sX + (ti * 16 + lrow) * SWP + k4 * 32 + koff;
            ua.h[0] = *(const v8bf*)pa; ua.h[1] = *(const v8bf*)(pa + 16);
            const bf16_t* pb = sX + (tj * 16 + lrow) * SWP + k4 * 32 + koff;
            ub.h[0] = *(const v8bf*)pb; ub.h[1] = *(const v8bf*)(pb + 16);
            c = __builtin_amdgcn_wmma_f32_16x16x32_bf16(false, ua.v, false, ub.v,
                                                        (short)0, c, false, false);
        }
        int nn = tj * 16 + lrow;
        for (int r = 0; r < 8; ++r) {
            int m = ti * 16 + r + koff;          // koff == 8*(lane>>4)
            og[rowBase[m] + nn] = c[r] * 0.08838834764831845f;   // 1/sqrt(128)
        }
    }
    // 16 diagonal tiles: predicate only here (nn > m)
    for (int d = wave; d < 16; d += 8) {
        v8f c = {};
        for (int k4 = 0; k4 < 4; ++k4) {
            FragBF ua;
            const bf16_t* pa = sX + (d * 16 + lrow) * SWP + k4 * 32 + koff;
            ua.h[0] = *(const v8bf*)pa; ua.h[1] = *(const v8bf*)(pa + 16);
            c = __builtin_amdgcn_wmma_f32_16x16x32_bf16(false, ua.v, false, ua.v,
                                                        (short)0, c, false, false);
        }
        int nn = d * 16 + lrow;
        for (int r = 0; r < 8; ++r) {
            int m = d * 16 + r + koff;
            if (nn > m)
                og[rowBase[m] + nn] = c[r] * 0.08838834764831845f;
        }
    }
}

// =========================================================================
extern "C" void kernel_launch(void* const* d_in, const int* in_sizes, int n_in,
                              void* d_out, int out_size, void* d_ws, size_t ws_size,
                              hipStream_t stream) {
    const int*   nf   = (const int*)d_in[0];
    const int*   eidx = (const int*)d_in[1];
    const int*   srcE = eidx;
    const int*   dstE = eidx + (size_t)NGRAPH * EPG;
    const float* emb  = (const float*)d_in[3];
    // params (jax pytree, dict keys sorted): embedding, exit{l1{b,w},l2{b,w},ln{b,g}},
    // layers[2]{gin_l1{b,w},gin_l2{b,w},gin_ln{b,g},post_l1{b,w},post_l2{b,w}}, norm{b,g}
    const float* ex_l1_b = (const float*)d_in[4];
    const float* ex_l1_w = (const float*)d_in[5];
    const float* ex_l2_b = (const float*)d_in[6];
    const float* ex_l2_w = (const float*)d_in[7];
    const float* ex_ln_b = (const float*)d_in[8];
    const float* ex_ln_g = (const float*)d_in[9];
    const float* norm_b  = (const float*)d_in[30];
    const float* norm_g  = (const float*)d_in[31];

    char* ws = (char*)d_ws;
    size_t off = 0;
    auto alloc = [&](size_t bytes) -> void* {
        void* p = ws + off; off += (bytes + 255) & ~(size_t)255; return p;
    };
    float*  X   = (float*)alloc((size_t)NTF * 4);
    float*  F   = (float*)alloc((size_t)NTF * 4);
    bf16_t* Hb  = (bf16_t*)alloc((size_t)NTF * 2);
    bf16_t* Kb  = (bf16_t*)alloc((size_t)NTF * 2);
    bf16_t* Wb[9];
    for (int i = 0; i < 9; ++i) Wb[i] = (bf16_t*)alloc(128 * 128 * 2);
    float*  means  = (float*)alloc(NGRAPH * 128 * 4);
    bf16_t* meansb = (bf16_t*)alloc(NGRAPH * 128 * 2);
    float*  exF    = (float*)alloc(NGRAPH * 128 * 4);
    float*  outF   = (float*)d_out;

    // convert weights to bf16: [l*4 + {gin1,gin2,post1,post2}], [8] = exit.l1
    const float* wsrc[9] = {
        (const float*)d_in[11], (const float*)d_in[13],
        (const float*)d_in[17], (const float*)d_in[19],
        (const float*)d_in[21], (const float*)d_in[23],
        (const float*)d_in[27], (const float*)d_in[29],
        ex_l1_w };
    for (int i = 0; i < 9; ++i)
        k_f2b<<<64, 256, 0, stream>>>(wsrc[i], Wb[i], 128 * 128);

    k_embed<<<NTF / 256, 256, 0, stream>>>(nf, emb, X);

    for (int l = 0; l < 2; ++l) {
        int pb = 10 + 10 * l;
        const float* gin1_b  = (const float*)d_in[pb + 0];
        const float* gin2_b  = (const float*)d_in[pb + 2];
        const float* ginln_b = (const float*)d_in[pb + 4];
        const float* ginln_g = (const float*)d_in[pb + 5];
        const float* post1_b = (const float*)d_in[pb + 6];
        const float* post2_b = (const float*)d_in[pb + 8];

        // h0 = x + agg  (bf16, LDS segment-sum)
        k_aggregate<<<NGRAPH, 1024, 0, stream>>>(X, srcE, dstE, Hb);
        // t1 = h0 @ W1 + b1                         -> F (f32)
        k_gemm128<0,1,0><<<NT / 256, 256, 0, stream>>>(Hb, Wb[l*4+0], gin1_b,
                                                       F, nullptr, NT);
        // r1 = relu(LN(t1))                         -> Hb (bf16)
        k_ln<0,1,0,1><<<NT / 8, 256, 0, stream>>>(F, nullptr, ginln_g, ginln_b,
                                                  nullptr, Hb, NT);
        // t2 = r1 @ W2 + b2                         -> Kb (bf16, feeds post MLP)
        k_gemm128<0,0,1><<<NT / 256, 256, 0, stream>>>(Hb, Wb[l*4+1], gin2_b,
                                                       nullptr, Kb, NT);
        // t3 = relu(t2 @ pW1 + pb1)                 -> Hb (bf16)
        k_gemm128<1,0,1><<<NT / 256, 256, 0, stream>>>(Kb, Wb[l*4+2], post1_b,
                                                       nullptr, Hb, NT);
        // t4 = t3 @ pW2 + pb2                       -> F (f32)
        k_gemm128<0,1,0><<<NT / 256, 256, 0, stream>>>(Hb, Wb[l*4+3], post2_b,
                                                       F, nullptr, NT);
        // x = LN(t4 (+ x if l>0))                   -> X (f32), Kb (bf16)
        if (l > 0)
            k_ln<1,0,1,1><<<NT / 8, 256, 0, stream>>>(F, X, norm_g, norm_b,
                                                      X, Kb, NT);
        else
            k_ln<0,0,1,1><<<NT / 8, 256, 0, stream>>>(F, nullptr, norm_g, norm_b,
                                                      X, Kb, NT);
    }

    // exit head
    k_meanpool<<<NGRAPH, 128, 0, stream>>>(X, means, meansb);
    k_gemm128<0,1,0><<<NGRAPH / 256, 256, 0, stream>>>(meansb, Wb[8], ex_l1_b,
                                                       exF, nullptr, NGRAPH);
    k_ln<0,0,1,0><<<NGRAPH / 8, 256, 0, stream>>>(exF, nullptr, ex_ln_g, ex_ln_b,
                                                  exF, nullptr, NGRAPH);
    k_exit_dot<<<NGRAPH / 8, 256, 0, stream>>>(exF, ex_l2_w, ex_l2_b, outF);

    // pairwise dot upper triangle (TDM-staged LDS)
    k_dots<<<NGRAPH, 256, 0, stream>>>(Kb, outF);
    (void)in_sizes; (void)n_in; (void)out_size; (void)ws_size;
}
